// GCNShallowRegression_29437705847601
// MI455X (gfx1250) — matrix operand order repaired and checked by reading, not compiled, not run on hardware
//
#include <hip/hip_runtime.h>

typedef __attribute__((ext_vector_type(16))) __bf16        v16bf;
typedef __attribute__((ext_vector_type(8)))  float         v8f;
typedef __attribute__((ext_vector_type(8)))  unsigned int  v8u;

// ---------------- helpers ----------------

__device__ __forceinline__ unsigned int f2bf_pk(float lo, float hi) {
  // round-to-nearest-even f32 -> bf16, packed pair (lo in [15:0], hi in [31:16])
  unsigned int a = __builtin_bit_cast(unsigned int, lo);
  unsigned int b = __builtin_bit_cast(unsigned int, hi);
  a = (a + 0x7FFFu + ((a >> 16) & 1u)) >> 16;
  b = (b + 0x7FFFu + ((b >> 16) & 1u)) & 0xFFFF0000u;
  return (a & 0xFFFFu) | b;
}

// Guaranteed-native non-returning FP32 atomic add at device scope: the RMW is
// performed by the L2 atomic units (S and deg stay L2-resident: 51MB << 192MB),
// tracked only by STOREcnt, no data return, no CAS loop.
__device__ __forceinline__ void gatomAddF(float* p, float v) {
  asm volatile("global_atomic_add_f32 %0, %1, off scope:SCOPE_DEV"
               :
               : "v"(p), "v"(v)
               : "memory");
}

// ---------------- kernels ----------------

// deg[i] = 1.0 (self loop)
__global__ __launch_bounds__(256) void k_fill_deg(float* __restrict__ deg, int n) {
  int i = blockIdx.x * 256 + threadIdx.x;
  if (i < n) deg[i] = 1.0f;
}

// deg[dst[e]] += 1
__global__ __launch_bounds__(256) void k_deg_edges(const int* __restrict__ dst,
                                                   float* __restrict__ deg, int e) {
  int i = blockIdx.x * 256 + threadIdx.x;
  if (i < e) gatomAddF(&deg[dst[i]], 1.0f);
}

// deg[i] <- rsqrt(deg[i])   (deg >= 1 always, no zero check needed)
__global__ __launch_bounds__(256) void k_rsqrt(float* __restrict__ deg, int n) {
  int i = blockIdx.x * 256 + threadIdx.x;
  if (i < n) deg[i] = rsqrtf(deg[i]);
}

// hs = (x @ W.T) * dinv[row]; S initialized to hs (self-loop message pre-scaled).
// One wave per 16-row stripe, all 128 output columns (8 WMMA tiles), K=128 in 4 steps.
__global__ __launch_bounds__(256) void k_gemm_hs(const float* __restrict__ x,
                                                 const float* __restrict__ W,
                                                 const float* __restrict__ dinv,
                                                 float* __restrict__ hs,
                                                 float* __restrict__ S, int n) {
  // W (128x128 f32 row-major) -> bf16 in LDS, row-major, packed pairs (32 KB)
  __shared__ unsigned int Wl[128 * 128 / 2];
  {
    const float4* W4 = (const float4*)W;
    for (int i = threadIdx.x; i < 4096; i += 256) {
      float4 w = W4[i];
      Wl[2 * i + 0] = f2bf_pk(w.x, w.y);
      Wl[2 * i + 1] = f2bf_pk(w.z, w.w);
    }
  }
  __syncthreads();

  const int lane = threadIdx.x & 31;
  const int wave = threadIdx.x >> 5;
  const int half = lane >> 4;   // 0: lanes 0-15, 1: lanes 16-31
  const int l16  = lane & 15;
  const int m0   = (blockIdx.x * 8 + wave) * 16;
  if (m0 >= n) return;          // uniform per wave (16 | N), EXEC stays all-ones

  v8f acc[8] = {};

  const float* xrow = x + (size_t)(m0 + l16) * 128;

  #pragma unroll
  for (int kk = 0; kk < 4; ++kk) {
    // A tile 16x32 bf16: lane holds K = kb..kb+15, contiguous, kb = kk*32 + half*16
    const int kb = kk * 32 + half * 16;
    float4 a0 = *(const float4*)(xrow + kb);
    float4 a1 = *(const float4*)(xrow + kb + 4);
    float4 a2 = *(const float4*)(xrow + kb + 8);
    float4 a3 = *(const float4*)(xrow + kb + 12);
    v8u au;
    au[0] = f2bf_pk(a0.x, a0.y); au[1] = f2bf_pk(a0.z, a0.w);
    au[2] = f2bf_pk(a1.x, a1.y); au[3] = f2bf_pk(a1.z, a1.w);
    au[4] = f2bf_pk(a2.x, a2.y); au[5] = f2bf_pk(a2.z, a2.w);
    au[6] = f2bf_pk(a3.x, a3.y); au[7] = f2bf_pk(a3.z, a3.w);
    v16bf A = __builtin_bit_cast(v16bf, au);

    #pragma unroll
    for (int t = 0; t < 8; ++t) {
      // B tile 32x16 bf16 (= rows of W, i.e. W.T column-major):
      // lane (col = t*16+l16) holds K = kk*32 + half*16 + (0..15), contiguous.
      const int col = t * 16 + l16;
      const int o   = (col * 128 + kk * 32 + half * 16) >> 1;  // uint index
      uint4 q0 = *(const uint4*)(&Wl[o]);
      uint4 q1 = *(const uint4*)(&Wl[o + 4]);
      v8u bu;
      bu[0] = q0.x; bu[1] = q0.y; bu[2] = q0.z; bu[3] = q0.w;
      bu[4] = q1.x; bu[5] = q1.y; bu[6] = q1.z; bu[7] = q1.w;
      v16bf B = __builtin_bit_cast(v16bf, bu);
      acc[t] = __builtin_amdgcn_wmma_f32_16x16x32_bf16(
          false, A, false, B, (short)0, acc[t], false, false);
    }
  }

  // C/D layout: lane<16 -> rows m0+0..7, lane>=16 -> rows m0+8..15; col = l16 per tile
  const int rbase = m0 + half * 8;
  float dv[8];
  #pragma unroll
  for (int r = 0; r < 8; ++r) dv[r] = dinv[rbase + r];
  #pragma unroll
  for (int t = 0; t < 8; ++t) {
    const int col = t * 16 + l16;
    #pragma unroll
    for (int r = 0; r < 8; ++r) {
      float v = acc[t][r] * dv[r];
      size_t idx = (size_t)(rbase + r) * 128 + col;
      hs[idx] = v;
      S[idx]  = v;
    }
  }
}

// S[dst] += hs[src]  (dinv[src] already folded into hs; dinv[dst] applied at the end)
// One wave per 16 consecutive edges, one at a time: coalesced 512B row gather, 128 f32 atomics into L2-resident S.
__global__ __launch_bounds__(256) void k_scatter(const int* __restrict__ src,
                                                 const int* __restrict__ dst,
                                                 const float* __restrict__ hs,
                                                 float* __restrict__ S, int e) {
  const int lane = threadIdx.x & 31;
  const int wave = threadIdx.x >> 5;
  const int e0   = (blockIdx.x * 8 + wave) * 16;
  for (int i = 0; i < 16; ++i) {
    int eid = e0 + i;
    if (eid >= e) break;
    int s = src[eid];
    int d = dst[eid];
    float4 v = ((const float4*)(hs + (size_t)s * 128))[lane];
    float* p = S + (size_t)d * 128 + lane * 4;
    gatomAddF(p + 0, v.x);
    gatomAddF(p + 1, v.y);
    gatomAddF(p + 2, v.z);
    gatomAddF(p + 3, v.w);
  }
}

// out[i] = sigmoid( relu(dinv[i]*S[i] + bias) . lin_w + lin_b ), one wave32 per node
__global__ __launch_bounds__(256) void k_final(const float* __restrict__ S,
                                               const float* __restrict__ dinv,
                                               const float* __restrict__ bias,
                                               const float* __restrict__ lin_w,
                                               const float* __restrict__ lin_b,
                                               float* __restrict__ out, int n) {
  const int lane = threadIdx.x & 31;
  const int wave = threadIdx.x >> 5;
  const int node = blockIdx.x * 8 + wave;
  if (node >= n) return;
  float di = dinv[node];
  float4 v = ((const float4*)(S + (size_t)node * 128))[lane];
  float4 b = ((const float4*)bias)[lane];
  float4 w = ((const float4*)lin_w)[lane];
  float ax = fmaxf(v.x * di + b.x, 0.0f);
  float ay = fmaxf(v.y * di + b.y, 0.0f);
  float az = fmaxf(v.z * di + b.z, 0.0f);
  float aw = fmaxf(v.w * di + b.w, 0.0f);
  float p = ax * w.x + ay * w.y + az * w.z + aw * w.w;
  #pragma unroll
  for (int off = 16; off > 0; off >>= 1) p += __shfl_down(p, off, 32);
  if (lane == 0) {
    float z = p + lin_b[0];
    out[node] = 1.0f / (1.0f + expf(-z));
  }
}

// ---------------- launch ----------------

extern "C" void kernel_launch(void* const* d_in, const int* in_sizes, int n_in,
                              void* d_out, int out_size, void* d_ws, size_t ws_size,
                              hipStream_t stream) {
  const float* x    = (const float*)d_in[0];
  const int*   ei   = (const int*)d_in[1];
  const float* W    = (const float*)d_in[2];
  const float* bias = (const float*)d_in[3];
  const float* lw   = (const float*)d_in[4];
  const float* lb   = (const float*)d_in[5];
  float*       out  = (float*)d_out;

  const int N = in_sizes[0] / 128;
  const int E = in_sizes[1] / 2;
  const int* src = ei;
  const int* dst = ei + E;

  // workspace carve-up: deg/dinv [N] | hs [N*128] | S [N*128]  (~103 MB total)
  float* deg = (float*)d_ws;
  size_t degPad = ((size_t)N + 63) & ~(size_t)63;
  float* hs = deg + degPad;
  float* S  = hs + (size_t)N * 128;

  k_fill_deg<<<(N + 255) / 256, 256, 0, stream>>>(deg, N);
  k_deg_edges<<<(E + 255) / 256, 256, 0, stream>>>(dst, deg, E);
  k_rsqrt<<<(N + 255) / 256, 256, 0, stream>>>(deg, N);
  k_gemm_hs<<<(N + 127) / 128, 256, 0, stream>>>(x, W, deg, hs, S, N);
  k_scatter<<<(E + 127) / 128, 256, 0, stream>>>(src, dst, hs, S, E);
  k_final<<<(N + 7) / 8, 256, 0, stream>>>(S, deg, bias, lw, lb, out, N);
}
